// QLSTM_83820581748791
// MI455X (gfx1250) — matrix operand order repaired and checked by reading, not compiled, not run on hardware
//
#include <hip/hip_runtime.h>

// ---------------------------------------------------------------------------
// QLSTM for MI455X (gfx1250): bf16 WMMA (v_wmma_f32_16x16x32_bf16) everywhere,
// TDM (tensor_load_to_lds) staging of the recurrent A-tile into LDS,
// ds_load_b128 A-operand reads, global_prefetch_b8 on B streams.
// Phases: pack (quaternion structure -> dense bf16, WMMA-B-swizzled),
//         gates GEMM (16384x4096, K=512),
//         256 sequential recurrent steps (64x4096, K=1024) with fused LSTM
//         pointwise update in registers,
//         output GEMM (8192x512, K=2048).
// ---------------------------------------------------------------------------

typedef __attribute__((ext_vector_type(16))) __bf16    v16bf;
typedef __attribute__((ext_vector_type(8)))  __bf16    v8bf;
typedef __attribute__((ext_vector_type(8)))  float     v8f;
typedef __attribute__((ext_vector_type(4)))  unsigned  v4u;
typedef __attribute__((ext_vector_type(8)))  int       v8i;
typedef __attribute__((ext_vector_type(4)))  int       v4i;

#define T_DIM 256
#define B_DIM 32
#define F_DIM 512
#define H_DIM 1024
#define O_DIM 512
#define BB    64      // 2*B (fwd + bwd batches)
#define NG    4096    // 4 gates * H

// ---------------------------------------------------------------------------
// Pack Xb = concat([x, x reversed in time], batch) as bf16 row-major (T*BB, F)
// ---------------------------------------------------------------------------
__global__ void pack_xb(const float* __restrict__ x, __bf16* __restrict__ xb) {
  int idx = blockIdx.x * blockDim.x + threadIdx.x;
  if (idx >= T_DIM * BB * F_DIM) return;
  int f   = idx % F_DIM;
  int row = idx / F_DIM;
  int b = row % BB, t = row / BB;
  float v = (b < B_DIM) ? x[(size_t)(t * B_DIM + b) * F_DIM + f]
                        : x[(size_t)((T_DIM - 1 - t) * B_DIM + (b - B_DIM)) * F_DIM + f];
  xb[idx] = (__bf16)v;
}

// Quaternion effective matrix element E[k][n] for gate-major N (4 gates * H).
__device__ __forceinline__ float quat_w(const float* __restrict__ W,
                                        int k, int n, int blockK) {
  const int   comp_tbl[4][4] = {{0,1,2,3},{1,0,3,2},{2,3,0,1},{3,2,1,0}};
  const float sign_tbl[4][4] = {{ 1.f, 1.f, 1.f, 1.f},
                                {-1.f, 1.f, 1.f,-1.f},
                                {-1.f,-1.f, 1.f, 1.f},
                                {-1.f, 1.f,-1.f, 1.f}};
  int g    = n >> 10;        // gate
  int nc   = n & 1023;
  int q    = nc >> 8;        // output component block (H/4 = 256)
  int ncol = nc & 255;
  int p    = k / blockK;     // input component block
  int kr   = k % blockK;
  int c    = comp_tbl[p][q];
  return sign_tbl[p][q] * W[((size_t)(g * 4 + c) * blockK + kr) * 256 + ncol];
}

// ---------------------------------------------------------------------------
// Pack quaternion weight into WMMA-B tile layout: tile (kt,nt) of 32x16,
// lane L holds col nt*16+(L&15), K rows kt*32+(L>>4)*8+j (j<8) / kt*32+16+(L>>4)*8+(j-8) (j>=8), 32B/lane.
// ---------------------------------------------------------------------------
__global__ void pack_wq(const float* __restrict__ W, __bf16* __restrict__ out,
                        int K, int blockK) {
  int idx = blockIdx.x * blockDim.x + threadIdx.x;
  if (idx >= K * NG) return;
  int tile = idx >> 9;
  int r    = idx & 511;
  int L = r >> 4, j = r & 15;
  int kt = tile >> 8;           // / (4096/16)
  int nt = tile & 255;
  int k = kt * 32 + (L >> 4) * 8 + (j < 8 ? j : 8 + j);
  int n = nt * 16 + (L & 15);
  out[idx] = (__bf16)quat_w(W, k, n, blockK);
}

// fco_w (2048 x 512) -> same B-swizzled layout (32 n-tiles)
__global__ void pack_fco(const float* __restrict__ W, __bf16* __restrict__ out) {
  int idx = blockIdx.x * blockDim.x + threadIdx.x;
  if (idx >= 2 * H_DIM * O_DIM) return;
  int tile = idx >> 9;
  int r    = idx & 511;
  int L = r >> 4, j = r & 15;
  int kt = tile >> 5;           // / (512/16)
  int nt = tile & 31;
  int k = kt * 32 + (L >> 4) * 8 + (j < 8 ? j : 8 + j);
  int n = nt * 16 + (L & 15);
  out[idx] = (__bf16)W[(size_t)k * O_DIM + n];
}

// bias_all = bx + bh, zero c and hs slot 0
__global__ void pack_bias_init(const float* __restrict__ bx,
                               const float* __restrict__ bh,
                               float* __restrict__ bias,
                               float* __restrict__ cbuf,
                               __bf16* __restrict__ hs0) {
  int idx = blockIdx.x * blockDim.x + threadIdx.x;
  if (idx < NG) bias[idx] = bx[idx] + bh[idx];
  if (idx < BB * H_DIM) { cbuf[idx] = 0.f; hs0[idx] = (__bf16)0.f; }
}

// ---------------------------------------------------------------------------
// Load A operand (16x32 bf16, wave32 layout) from a row-major row pointer.
// ---------------------------------------------------------------------------
__device__ __forceinline__ v16bf load_a(const __bf16* __restrict__ p) {
  v8bf lo = *(const v8bf*)p;
  v8bf hi = *(const v8bf*)(p + 16);
  v16bf a;
#pragma unroll
  for (int i = 0; i < 8; ++i) { a[i] = lo[i]; a[8 + i] = hi[i]; }
  return a;
}

// ---------------------------------------------------------------------------
// gates = Xb @ Wx_eff + (bx+bh), bf16 out, row-major (T*BB, 4096)
// ---------------------------------------------------------------------------
__global__ void gates_gemm(const __bf16* __restrict__ Xb,
                           const __bf16* __restrict__ Wp,
                           const float* __restrict__ bias,
                           __bf16* __restrict__ gates) {
  int lane  = threadIdx.x & 31;
  int gw    = blockIdx.x * (blockDim.x >> 5) + (threadIdx.x >> 5);
  int mtile = gw >> 8;    // 1024 m-tiles
  int ntile = gw & 255;   // 256 n-tiles
  int arow  = mtile * 16 + (lane & 15);
  int akoff = (lane >> 4) * 8;
  v8f acc = {};
#pragma unroll
  for (int kk = 0; kk < F_DIM; kk += 32) {
    const __bf16* ap = Xb + (size_t)arow * F_DIM + kk + akoff;
    v16bf a = load_a(ap);
    const __bf16* bp = Wp + ((size_t)((kk >> 5) * 256 + ntile) * 512 + lane * 16);
    __builtin_prefetch(bp + 256 * 512, 0, 1);   // next K-tile -> global_prefetch
    v16bf b = *(const v16bf*)bp;
    acc = __builtin_amdgcn_wmma_f32_16x16x32_bf16(false, a, false, b,
                                                  (short)0, acc, false, false);
  }
  int n = ntile * 16 + (lane & 15);
  float bv = bias[n];
  int mbase = (lane < 16) ? 0 : 8;
#pragma unroll
  for (int r = 0; r < 8; ++r) {
    int m = mbase + r;
    gates[(size_t)(mtile * 16 + m) * NG + n] = (__bf16)(acc[r] + bv);
  }
}

// ---------------------------------------------------------------------------
// One recurrent step. Block = 4 waves sharing one 16(batch)x1024 A-tile,
// staged global->LDS by the Tensor Data Mover (one DMA per block), then each
// wave computes one 16x16 hidden tile across all 4 gates (K=1024, 128 WMMAs)
// and performs the fused LSTM pointwise update in registers.
// hs has T+1 slots; slot t = h entering step t.
// ---------------------------------------------------------------------------
__global__ void __launch_bounds__(128) scan_step(const __bf16* __restrict__ Whp,
                                                 const __bf16* __restrict__ gates,
                                                 __bf16* __restrict__ hs,
                                                 float* __restrict__ cbuf, int t) {
  __shared__ __bf16 aTile[16 * H_DIM];          // 32 KB
  int lane = threadIdx.x & 31;
  int wv   = threadIdx.x >> 5;
  int mt   = blockIdx.x >> 4;                   // 4 batch tiles
  int nb   = (blockIdx.x & 15) * 4 + wv;        // 64 hidden tiles

  const __bf16* hprev = hs + (size_t)t * BB * H_DIM + (size_t)mt * 16 * H_DIM;

  if (wv == 0) {
    // --- Tensor DMA descriptor (D#): 2-D tile, 16 rows x 1024 cols, 2B elems
    unsigned lds_off = (unsigned)(size_t)(void*)aTile;
    unsigned long long ga = (unsigned long long)(const void*)hprev;
    v4u g0;
    g0[0] = 1u;                                    // count=1 (valid), user mode
    g0[1] = lds_off;                               // lds_addr (bytes)
    g0[2] = (unsigned)(ga & 0xFFFFFFFFull);        // global_addr[31:0]
    g0[3] = (unsigned)((ga >> 32) & 0x01FFFFFFull) // global_addr[56:32]
            | (2u << 30);                          // type=2 ("image")
    v8i g1;
    g1[0] = 1 << 16;                               // workgroup_mask=0, data_size=2B
    g1[1] = (int)(((unsigned)H_DIM & 0xFFFFu) << 16);  // tensor_dim0[15:0]=1024
    g1[2] = (int)((16u & 0xFFFFu) << 16);          // td0 hi=0 | tensor_dim1[15:0]=16
    g1[3] = (int)(((unsigned)H_DIM & 0xFFFFu) << 16);  // td1 hi=0 | tile_dim0=1024
    g1[4] = 16;                                    // tile_dim1=16, tile_dim2=0
    g1[5] = H_DIM;                                 // tensor_dim0_stride[31:0]=1024
    g1[6] = 0;                                     // stride0 hi, stride1 lo
    g1[7] = 0;
    v4i gz4 = {0, 0, 0, 0};
    v8i gz8 = {0, 0, 0, 0, 0, 0, 0, 0};
    // 6-arg (clang-23 / therock) form of the TDM builtin
    __builtin_amdgcn_tensor_load_to_lds(g0, g1, gz4, gz4, gz8, 0);
    __builtin_amdgcn_s_wait_tensorcnt(0);
  }
  __syncthreads();

  int arow_l = (lane & 15);                       // row within LDS tile
  int akoff  = (lane >> 4) * 8;
  v8f acc0 = {}, acc1 = {}, acc2 = {}, acc3 = {};
#pragma unroll 8
  for (int kk = 0; kk < H_DIM; kk += 32) {
    v16bf a = load_a(&aTile[(size_t)arow_l * H_DIM + kk + akoff]);   // ds_load_b128
    int kt = kk >> 5;
    const __bf16* b0p = Whp + ((size_t)(kt * 256 + 0 * 64 + nb) * 512 + lane * 16);
    __builtin_prefetch(b0p + 256 * 512, 0, 1);
    v16bf b0 = *(const v16bf*)b0p;
    v16bf b1 = *(const v16bf*)(Whp + ((size_t)(kt * 256 + 1 * 64 + nb) * 512 + lane * 16));
    v16bf b2 = *(const v16bf*)(Whp + ((size_t)(kt * 256 + 2 * 64 + nb) * 512 + lane * 16));
    v16bf b3 = *(const v16bf*)(Whp + ((size_t)(kt * 256 + 3 * 64 + nb) * 512 + lane * 16));
    acc0 = __builtin_amdgcn_wmma_f32_16x16x32_bf16(false, a, false, b0, (short)0, acc0, false, false);
    acc1 = __builtin_amdgcn_wmma_f32_16x16x32_bf16(false, a, false, b1, (short)0, acc1, false, false);
    acc2 = __builtin_amdgcn_wmma_f32_16x16x32_bf16(false, a, false, b2, (short)0, acc2, false, false);
    acc3 = __builtin_amdgcn_wmma_f32_16x16x32_bf16(false, a, false, b3, (short)0, acc3, false, false);
  }

  __bf16* hnext = hs + (size_t)(t + 1) * BB * H_DIM;
  int n = lane & 15;
  int mbase = (lane < 16) ? 0 : 8;
#pragma unroll
  for (int r = 0; r < 8; ++r) {
    int m = mbase + r;
    int b = mt * 16 + m;
    int h = nb * 16 + n;
    size_t grow = ((size_t)t * BB + b) * NG;
    float pf = acc0[r] + (float)gates[grow + 0 * H_DIM + h];
    float pi = acc1[r] + (float)gates[grow + 1 * H_DIM + h];
    float po = acc2[r] + (float)gates[grow + 2 * H_DIM + h];
    float pa = acc3[r] + (float)gates[grow + 3 * H_DIM + h];
    float ft = 1.f / (1.f + __expf(-pf));
    float it = 1.f / (1.f + __expf(-pi));
    float ot = 1.f / (1.f + __expf(-po));
    size_t ci = (size_t)b * H_DIM + h;
    float cn = it * tanhf(pa) + ft * cbuf[ci];
    cbuf[ci] = cn;
    hnext[ci] = (__bf16)(ot * tanhf(cn));
  }
}

// ---------------------------------------------------------------------------
// out = concat(h_f, h_b) @ fco_w + fco_b  -> (T*B, O) fp32
// ---------------------------------------------------------------------------
__global__ void out_gemm(const __bf16* __restrict__ hs,
                         const __bf16* __restrict__ Fp,
                         const float* __restrict__ fco_b,
                         float* __restrict__ out) {
  int lane  = threadIdx.x & 31;
  int gw    = blockIdx.x * (blockDim.x >> 5) + (threadIdx.x >> 5);
  int mtile = gw >> 5;    // 512 m-tiles (8192 rows)
  int ntile = gw & 31;    // 32 n-tiles
  int r0 = mtile * 16 + (lane & 15);
  int t = r0 >> 5;        // 32 rows per timestep -> tile stays within one t
  int b = r0 & 31;
  int akoff = (lane >> 4) * 8;
  const __bf16* af = hs + ((size_t)(t + 1) * BB + b) * H_DIM;
  const __bf16* ab = hs + ((size_t)(T_DIM - t) * BB + B_DIM + b) * H_DIM;
  v8f acc = {};
#pragma unroll 4
  for (int kk = 0; kk < 2 * H_DIM; kk += 32) {
    const __bf16* ap = (kk < H_DIM) ? (af + kk + akoff) : (ab + (kk - H_DIM) + akoff);
    v16bf a = load_a(ap);
    const __bf16* bp = Fp + ((size_t)((kk >> 5) * 32 + ntile) * 512 + lane * 16);
    __builtin_prefetch(bp + 32 * 512, 0, 1);
    v16bf bv = *(const v16bf*)bp;
    acc = __builtin_amdgcn_wmma_f32_16x16x32_bf16(false, a, false, bv,
                                                  (short)0, acc, false, false);
  }
  int n = ntile * 16 + (lane & 15);
  float bb = fco_b[n];
  int mbase = (lane < 16) ? 0 : 8;
#pragma unroll
  for (int r = 0; r < 8; ++r) {
    int m = mbase + r;
    out[(size_t)(mtile * 16 + m) * O_DIM + n] = acc[r] + bb;
  }
}

// ---------------------------------------------------------------------------
extern "C" void kernel_launch(void* const* d_in, const int* in_sizes, int n_in,
                              void* d_out, int out_size, void* d_ws, size_t ws_size,
                              hipStream_t stream) {
  const float* x  = (const float*)d_in[0];
  const float* Wx = (const float*)d_in[1];
  const float* bx = (const float*)d_in[2];
  const float* Wh = (const float*)d_in[3];
  const float* bh = (const float*)d_in[4];
  const float* fw = (const float*)d_in[5];
  const float* fb = (const float*)d_in[6];
  float* out = (float*)d_out;

  char* ws = (char*)d_ws;
  size_t off = 0;
  __bf16* XB  = (__bf16*)(ws + off); off += (size_t)T_DIM * BB * F_DIM * 2;        // 16 MB
  __bf16* WXP = (__bf16*)(ws + off); off += (size_t)F_DIM * NG * 2;                // 4 MB
  __bf16* WHP = (__bf16*)(ws + off); off += (size_t)H_DIM * NG * 2;                // 8 MB
  __bf16* FCP = (__bf16*)(ws + off); off += (size_t)2 * H_DIM * O_DIM * 2;         // 2 MB
  float*  BIAS= (float*) (ws + off); off += (size_t)NG * 4;
  float*  CBUF= (float*) (ws + off); off += (size_t)BB * H_DIM * 4;
  __bf16* HS  = (__bf16*)(ws + off); off += (size_t)(T_DIM + 1) * BB * H_DIM * 2;  // 33.5 MB
  __bf16* GATES=(__bf16*)(ws + off); off += (size_t)T_DIM * BB * NG * 2;           // 134 MB

  // --- pack / init ---
  pack_xb<<<(T_DIM * BB * F_DIM + 255) / 256, 256, 0, stream>>>(x, XB);
  pack_wq<<<(F_DIM * NG + 255) / 256, 256, 0, stream>>>(Wx, WXP, F_DIM, F_DIM / 4);
  pack_wq<<<(H_DIM * NG + 255) / 256, 256, 0, stream>>>(Wh, WHP, H_DIM, H_DIM / 4);
  pack_fco<<<(2 * H_DIM * O_DIM + 255) / 256, 256, 0, stream>>>(fw, FCP);
  pack_bias_init<<<(BB * H_DIM + 255) / 256, 256, 0, stream>>>(bx, bh, BIAS, CBUF, HS);

  // --- input projection: (16384 x 4096) = Xb @ Wx_eff ---
  gates_gemm<<<(1024 * 256) / 4, 128, 0, stream>>>(XB, WXP, BIAS, GATES);

  // --- sequential recurrence: 64 blocks x 4 waves per step ---
  for (int t = 0; t < T_DIM; ++t)
    scan_step<<<4 * 16, 128, 0, stream>>>(WHP, GATES, HS, CBUF, t);

  // --- output projection: (8192 x 512) ---
  out_gemm<<<(512 * 32) / 4, 128, 0, stream>>>(HS, FCP, fb, out);
}
